// VisNetEncoder_84524956385315
// MI455X (gfx1250) — matrix operand order repaired and checked by reading, not compiled, not run on hardware
//
#include <hip/hip_runtime.h>
#include <math.h>

// ---------------------------------------------------------------------------
// ViSNet encoder for MI455X (gfx1250, wave32, WMMA f32_16x16x32_f16)
// GEMM tiles staged in LDS in fragment-ready order -> ds_load_b128 fragment
// reads instead of per-half ds_load_u16 gathers.
// ---------------------------------------------------------------------------

#define NNODE   4096
#define NEDGE   65536
#define HDIM    256
#define NHEADS  8
#define HDHEAD  32
#define NRBF    32
#define NLAYER  6
#define CUTOFFR 5.0f

typedef __attribute__((ext_vector_type(16))) _Float16 v16h;
typedef __attribute__((ext_vector_type(8)))  _Float16 v8h;
typedef __attribute__((ext_vector_type(8)))  float    v8f;

static __device__ __forceinline__ float silu_f(float x) {
    return x / (1.0f + __expf(-x));
}

// ---------------------------------------------------------------------------
// GEMM: C[M x Nc] = act(A[M x K] @ B[K x Nc] + bias)
// Requirements: M % 128 == 0, Nc % 64 == 0, K % 32 == 0 (true for all calls).
// Block = 256 threads (8 waves). Tile = 128x64, K step 32.
// Wave w owns rows [w*16, w*16+16) x 64 cols (4 wmma accumulators).
// LDS tiles are stored in WMMA fragment order:
//   Afrag[rowgrp][lane][j]: lane = (k-half)*16 + (m&15);
//     j=0..15 <-> k = {0..7,16..23} (lanes 0-15) / {8..15,24..31} (lanes 16-31)
//   Bfrag[coltile][lane][j]: same k order as Afrag; lane = (k-half)*16 + (n&15)
// so each lane's fragment is 16 contiguous halves (2x ds_load_b128).
// ---------------------------------------------------------------------------
template <int ACT> // 0 = none, 1 = silu
__global__ __launch_bounds__(256)
void gemm_wmma_kernel(const float* __restrict__ A, const float* __restrict__ B,
                      const float* __restrict__ bias, float* __restrict__ Cmat,
                      int M, int K, int Nc) {
    __shared__ __align__(32) _Float16 Afrag[8][32][16]; // 8 KB
    __shared__ __align__(32) _Float16 Bfrag[4][32][16]; // 4 KB

    const int tid  = threadIdx.x;
    const int wave = tid >> 5;
    const int lane = tid & 31;
    const int laneHalf = lane >> 4; // 0: lanes 0-15, 1: lanes 16-31
    const int lmod = lane & 15;
    const int rowBase = blockIdx.x * 128;
    const int colBase = blockIdx.y * 64;

    v8f acc[4];
    #pragma unroll
    for (int nt = 0; nt < 4; ++nt) {
        float bv = bias ? bias[colBase + nt * 16 + lmod] : 0.0f;
        #pragma unroll
        for (int r = 0; r < 8; ++r) acc[nt][r] = bv;
    }

    for (int k0 = 0; k0 < K; k0 += 32) {
        // ---- stage A tile (128x32): 512 chunks of 8 k-values, 2 per thread
        #pragma unroll
        for (int i = 0; i < 2; ++i) {
            int c  = i * 256 + tid;      // 0..511
            int rr = c >> 2;             // 0..127 row in tile
            int kc = (c & 3) * 8;        // 0,8,16,24
            const float* ap = A + (long long)(rowBase + rr) * K + (k0 + kc);
            float4 f0 = *(const float4*)ap;
            float4 f1 = *(const float4*)(ap + 4);
            if (k0 + 32 < K) __builtin_prefetch(ap + 32, 0, 3);
            v8h hv;
            hv[0] = (_Float16)f0.x; hv[1] = (_Float16)f0.y;
            hv[2] = (_Float16)f0.z; hv[3] = (_Float16)f0.w;
            hv[4] = (_Float16)f1.x; hv[5] = (_Float16)f1.y;
            hv[6] = (_Float16)f1.z; hv[7] = (_Float16)f1.w;
            int rowgrp = rr >> 4;
            int m      = rr & 15;
            int lh     = (kc >> 3) & 1;          // k-half select
            int j0     = (kc & 16) >> 1;         // 0 or 8
            *(v8h*)&Afrag[rowgrp][lh * 16 + m][j0] = hv;
        }
        // ---- stage B tile (32x64): 256 chunks of 8 k-values, 1 per thread
        {
            int cc = tid & 63;                   // column in tile
            int kc = (tid >> 6) * 8;             // 0,8,16,24
            const float* bp = B + (long long)(k0 + kc) * Nc + (colBase + cc);
            v8h hv;
            #pragma unroll
            for (int jj = 0; jj < 8; ++jj)
                hv[jj] = (_Float16)bp[(long long)jj * Nc];
            int nt    = cc >> 4;
            int lane2 = ((kc >> 3) & 1) * 16 + (cc & 15);
            int j0    = (kc & 16) >> 1;
            *(v8h*)&Bfrag[nt][lane2][j0] = hv;
        }
        __syncthreads();

        v16h af = *(const v16h*)&Afrag[wave][lane][0];
        #pragma unroll
        for (int nt = 0; nt < 4; ++nt) {
            v16h bf = *(const v16h*)&Bfrag[nt][lane][0];
            acc[nt] = __builtin_amdgcn_wmma_f32_16x16x32_f16(
                false, af, false, bf, (short)0, acc[nt], false, false);
        }
        __syncthreads();
    }

    // epilogue: D f32 16x16 layout: row m = r + (lane>=16)*8, col n = lane&15
    const int mrow0 = rowBase + wave * 16 + laneHalf * 8;
    #pragma unroll
    for (int nt = 0; nt < 4; ++nt) {
        const int n = colBase + nt * 16 + lmod;
        #pragma unroll
        for (int r = 0; r < 8; ++r) {
            float vv = acc[nt][r];
            if (ACT == 1) vv = silu_f(vv);
            Cmat[(long long)(mrow0 + r) * Nc + n] = vv;
        }
    }
}

// ---------------------------------------------------------------------------
// Edge geometry: r, cosine cutoff, mask, unit vectors, ExpNorm RBF
// ---------------------------------------------------------------------------
__global__ void edge_geom_kernel(const float* __restrict__ pos, const int* __restrict__ ei,
                                 float* __restrict__ r, float* __restrict__ Cc,
                                 float* __restrict__ mask, float* __restrict__ dij,
                                 float* __restrict__ earbf) {
    int e = blockIdx.x * blockDim.x + threadIdx.x;
    if (e >= NEDGE) return;
    int s = ei[e], d = ei[NEDGE + e];
    float ex = pos[s * 3 + 0] - pos[d * 3 + 0];
    float ey = pos[s * 3 + 1] - pos[d * 3 + 1];
    float ez = pos[s * 3 + 2] - pos[d * 3 + 2];
    float m = (s != d) ? 1.0f : 0.0f;
    float ss = ex * ex + ey * ey + ez * ez;
    float rs = sqrtf((s != d) ? ss : 1.0f);
    float rv = rs * m;
    r[e] = rv;
    mask[e] = m;
    float inv = m / rs;
    dij[e * 3 + 0] = ex * inv;
    dij[e * 3 + 1] = ey * inv;
    dij[e * 3 + 2] = ez * inv;
    float Cv = (rv < CUTOFFR) ? 0.5f * (cosf(rv * (3.14159265358979323846f / CUTOFFR)) + 1.0f)
                              : 0.0f;
    Cc[e] = Cv;
    const float start = expf(-CUTOFFR);
    const float betaq = (2.0f / (float)NRBF) * (1.0f - start);
    const float beta = 1.0f / (betaq * betaq);
    float eprev = expf(-rv); // alpha = 5/CUTOFF = 1
    #pragma unroll
    for (int i = 0; i < NRBF; ++i) {
        float mean = start + (1.0f - start) * ((float)i / (float)(NRBF - 1));
        float t = eprev - mean;
        earbf[(long long)e * NRBF + i] = Cv * __expf(-beta * t * t);
    }
}

// ---------------------------------------------------------------------------
// NeighborEmbedding aggregation: agg[dst,h] += nbr_emb[z[src],h]*W[e,h]*(C*mask)
// ---------------------------------------------------------------------------
__global__ void neighbor_agg_kernel(const float* __restrict__ W, const float* __restrict__ Cc,
                                    const float* __restrict__ mask, const int* __restrict__ ei,
                                    const int* __restrict__ z, const float* __restrict__ nbr_emb,
                                    float* __restrict__ agg) {
    long long i = (long long)blockIdx.x * blockDim.x + threadIdx.x;
    if (i >= (long long)NEDGE * HDIM) return;
    int e = (int)(i >> 8), h = (int)(i & 255);
    int s = ei[e], d = ei[NEDGE + e];
    float w = W[i] * Cc[e] * mask[e];
    atomicAdd(&agg[(long long)d * HDIM + h], nbr_emb[(long long)z[s] * HDIM + h] * w);
}

__global__ void xcat_kernel(const int* __restrict__ z, const float* __restrict__ emb,
                            const float* __restrict__ agg, float* __restrict__ xcat) {
    long long i = (long long)blockIdx.x * blockDim.x + threadIdx.x;
    if (i >= (long long)NNODE * HDIM) return;
    int n = (int)(i >> 8), h = (int)(i & 255);
    xcat[(long long)n * 512 + h]       = emb[(long long)z[n] * HDIM + h];
    xcat[(long long)n * 512 + 256 + h] = agg[i];
}

__global__ void eatt_init_kernel(const float* __restrict__ x, const int* __restrict__ ei,
                                 const float* __restrict__ eproj, float* __restrict__ eatt) {
    long long i = (long long)blockIdx.x * blockDim.x + threadIdx.x;
    if (i >= (long long)NEDGE * HDIM) return;
    int e = (int)(i >> 8), h = (int)(i & 255);
    int s = ei[e], d = ei[NEDGE + e];
    eatt[i] = (x[(long long)s * HDIM + h] + x[(long long)d * HDIM + h]) * eproj[i];
}

// ---------------------------------------------------------------------------
// LayerNorm over H=256 per row: one wave32 per row, 8 rows per block
// ---------------------------------------------------------------------------
__global__ __launch_bounds__(256)
void layernorm_kernel(const float* __restrict__ x, const float* __restrict__ g,
                      const float* __restrict__ b, float* __restrict__ y, int M) {
    int row = blockIdx.x * 8 + (threadIdx.x >> 5);
    int lane = threadIdx.x & 31;
    if (row >= M) return;
    const float* xr = x + (long long)row * HDIM;
    float s = 0.0f, s2 = 0.0f;
    #pragma unroll
    for (int i = 0; i < 8; ++i) {
        float v = xr[lane + 32 * i];
        s += v; s2 += v * v;
    }
    #pragma unroll
    for (int off = 16; off > 0; off >>= 1) {
        s  += __shfl_xor(s, off, 32);
        s2 += __shfl_xor(s2, off, 32);
    }
    float mu = s * (1.0f / HDIM);
    float var = s2 * (1.0f / HDIM) - mu * mu;
    float rstd = rsqrtf(var + 1e-5f);
    float* yr = y + (long long)row * HDIM;
    #pragma unroll
    for (int i = 0; i < 8; ++i) {
        int h = lane + 32 * i;
        yr[h] = (xr[h] - mu) * rstd * g[h] + b[h];
    }
}

__global__ void vecn_kernel(const float* __restrict__ vec, const float* __restrict__ vln,
                            float* __restrict__ vecn) {
    long long i = (long long)blockIdx.x * blockDim.x + threadIdx.x;
    if (i >= (long long)NNODE * 3 * HDIM) return;
    vecn[i] = vec[i] * vln[i & 255];
}

// vec_dot[n,h] = sum_axis vecp[n,axis,h] * vecp[n,axis,H+h]
__global__ void vecdot_kernel(const float* __restrict__ vecp, float* __restrict__ vdot) {
    long long i = (long long)blockIdx.x * blockDim.x + threadIdx.x;
    if (i >= (long long)NNODE * HDIM) return;
    int n = (int)(i >> 8), h = (int)(i & 255);
    const float* base = vecp + (long long)n * 2304;
    float a = 0.0f;
    #pragma unroll
    for (int ax = 0; ax < 3; ++ax)
        a += base[ax * 768 + h] * base[ax * 768 + 256 + h];
    vdot[i] = a;
}

// attn[e,head] = silu(sum_hd q[dst]*k[src]*dk[e]) * C[e]   (float4 inner loop)
__global__ void attn_kernel(const float* __restrict__ q, const float* __restrict__ k,
                            const float* __restrict__ dk, const float* __restrict__ Cc,
                            const int* __restrict__ ei, float* __restrict__ attn) {
    long long i = (long long)blockIdx.x * blockDim.x + threadIdx.x;
    if (i >= (long long)NEDGE * NHEADS) return;
    int e = (int)(i >> 3), hh = (int)(i & 7);
    int s = ei[e], d = ei[NEDGE + e];
    const float4* qp = (const float4*)(q  + (long long)d * HDIM + hh * HDHEAD);
    const float4* kp = (const float4*)(k  + (long long)s * HDIM + hh * HDHEAD);
    const float4* dp = (const float4*)(dk + (long long)e * HDIM + hh * HDHEAD);
    float acc = 0.0f;
    #pragma unroll
    for (int j = 0; j < HDHEAD / 4; ++j) {
        float4 a = qp[j], b = kp[j], c = dp[j];
        acc += a.x * b.x * c.x + a.y * b.y * c.y + a.z * b.z * c.z + a.w * b.w * c.w;
    }
    attn[i] = silu_f(acc) * Cc[e];
}

// vm[e,h] = v[src,h] * dv[e,h] * attn[e, h/32]
__global__ void vm_kernel(const float* __restrict__ v, const float* __restrict__ dv,
                          const float* __restrict__ attn, const int* __restrict__ ei,
                          float* __restrict__ vm) {
    long long i = (long long)blockIdx.x * blockDim.x + threadIdx.x;
    if (i >= (long long)NEDGE * HDIM) return;
    int e = (int)(i >> 8), h = (int)(i & 255);
    int s = ei[e];
    vm[i] = v[(long long)s * HDIM + h] * dv[i] * attn[(long long)e * NHEADS + (h >> 5)];
}

// scatter: x_agg[dst] += vm ; vec_agg[dst,ax] += vecn[src,ax]*s1 + s2*d_ij[ax]
__global__ void scatter_kernel(const float* __restrict__ vm, const float* __restrict__ sbuf,
                               const float* __restrict__ vecn, const float* __restrict__ dij,
                               const int* __restrict__ ei, float* __restrict__ xagg,
                               float* __restrict__ vagg) {
    long long i = (long long)blockIdx.x * blockDim.x + threadIdx.x;
    if (i >= (long long)NEDGE * HDIM) return;
    int e = (int)(i >> 8), h = (int)(i & 255);
    int s = ei[e], d = ei[NEDGE + e];
    atomicAdd(&xagg[(long long)d * HDIM + h], vm[i]);
    float s1 = sbuf[(long long)e * 512 + h];
    float s2 = sbuf[(long long)e * 512 + 256 + h];
    #pragma unroll
    for (int ax = 0; ax < 3; ++ax) {
        float msg = vecn[((long long)s * 3 + ax) * HDIM + h] * s1 + s2 * dij[e * 3 + ax];
        atomicAdd(&vagg[((long long)d * 3 + ax) * HDIM + h], msg);
    }
}

// x += vec_dot*o2 + o3 ; vec += vec3*o1 + vec_agg
__global__ void update_kernel(const float* __restrict__ o, const float* __restrict__ vdot,
                              const float* __restrict__ vecp, const float* __restrict__ vagg,
                              float* __restrict__ x, float* __restrict__ vec) {
    long long i = (long long)blockIdx.x * blockDim.x + threadIdx.x;
    if (i >= (long long)NNODE * HDIM) return;
    int n = (int)(i >> 8), h = (int)(i & 255);
    float o1 = o[(long long)n * 768 + h];
    float o2 = o[(long long)n * 768 + 256 + h];
    float o3 = o[(long long)n * 768 + 512 + h];
    x[i] += vdot[i] * o2 + o3;
    #pragma unroll
    for (int ax = 0; ax < 3; ++ax) {
        long long vi = ((long long)n * 3 + ax) * HDIM + h;
        vec[vi] += vecp[(long long)n * 2304 + ax * 768 + 512 + h] * o1 + vagg[vi];
    }
}

// w_dot / t_dot per edge: sum over spatial axis of projected vecn at dst/src
__global__ void dots_kernel(const float* __restrict__ wt, const int* __restrict__ ei,
                            float* __restrict__ wdot, float* __restrict__ tdot) {
    long long i = (long long)blockIdx.x * blockDim.x + threadIdx.x;
    if (i >= (long long)NEDGE * HDIM) return;
    int e = (int)(i >> 8), h = (int)(i & 255);
    int s = ei[e], d = ei[NEDGE + e];
    const long long CH = (long long)3 * NNODE * HDIM; // per-projection chunk
    float wd = 0.0f, td = 0.0f;
    #pragma unroll
    for (int ax = 0; ax < 3; ++ax) {
        long long si = ((long long)s * 3 + ax) * HDIM + h;
        long long di = ((long long)d * 3 + ax) * HDIM + h;
        wd += wt[1 * CH + di] * wt[0 * CH + si];  // w_trg[dst] * w_src[src]
        td += wt[3 * CH + di] * wt[2 * CH + si];  // t_trg[dst] * t_src[src]
    }
    wdot[i] = wd;
    tdot[i] = td;
}

__global__ void eatt_update_kernel(const float* __restrict__ f, const float* __restrict__ wdot,
                                   const float* __restrict__ tdot, float* __restrict__ eatt) {
    long long i = (long long)blockIdx.x * blockDim.x + threadIdx.x;
    if (i >= (long long)NEDGE * HDIM) return;
    int e = (int)(i >> 8), h = (int)(i & 255);
    eatt[i] += f[(long long)e * 512 + h] * wdot[i] + f[(long long)e * 512 + 256 + h] * tdot[i];
}

__global__ void vout_kernel(const float* __restrict__ vec, const float* __restrict__ voutw,
                            float* __restrict__ out) {
    long long i = (long long)blockIdx.x * blockDim.x + threadIdx.x;
    if (i >= (long long)NNODE * 3 * HDIM) return;
    out[(long long)NNODE * HDIM + i] = vec[i] * voutw[i & 255];
}

// ---------------------------------------------------------------------------
// Host orchestration
// ---------------------------------------------------------------------------
extern "C" void kernel_launch(void* const* d_in, const int* in_sizes, int n_in,
                              void* d_out, int out_size, void* d_ws, size_t ws_size,
                              hipStream_t stream) {
    const int*   z      = (const int*)d_in[0];
    const float* pos    = (const float*)d_in[1];
    const int*   ei     = (const int*)d_in[3];
    const float* emb    = (const float*)d_in[4];
    const float* nbr    = (const float*)d_in[5];
    const float* dist_w = (const float*)d_in[6];
    const float* dist_b = (const float*)d_in[7];
    const float* comb_w = (const float*)d_in[8];
    const float* comb_b = (const float*)d_in[9];
    const float* edge_w = (const float*)d_in[10];
    const float* edge_b = (const float*)d_in[11];
    const float* ln_g   = (const float*)d_in[12];
    const float* ln_b   = (const float*)d_in[13];
    const float* vln_w  = (const float*)d_in[14];
    const float* vpw    = (const float*)d_in[15];
    const float* aw     = (const float*)d_in[16];
    const float* ab     = (const float*)d_in[17];
    const float* sw     = (const float*)d_in[18];
    const float* sb     = (const float*)d_in[19];
    const float* ow     = (const float*)d_in[20];
    const float* ob     = (const float*)d_in[21];
    const float* fw     = (const float*)d_in[22];
    const float* fb     = (const float*)d_in[23];
    const float* wtw    = (const float*)d_in[24];
    const float* outg   = (const float*)d_in[25];
    const float* outb   = (const float*)d_in[26];
    const float* voutw  = (const float*)d_in[27];
    (void)in_sizes; (void)n_in; (void)out_size; (void)ws_size;

    float* ws  = (float*)d_ws;
    float* out = (float*)d_out;

    const long long E = NEDGE, Nn = NNODE;
    // edge-scale buffers
    float* r     = ws;
    float* Cc    = ws + 1 * E;
    float* mask  = ws + 2 * E;
    float* dij   = ws + 3 * E;    // 3E
    float* earbf = ws + 6 * E;    // 32E
    float* eatt  = ws + 38 * E;   // 256E
    float* be1   = ws + 294 * E;  // 256E (dk / w_dot)
    float* be2   = ws + 550 * E;  // 256E (dv / t_dot)
    float* be3   = ws + 806 * E;  // 512E (W / eproj / s / f)
    float* be4   = ws + 1318 * E; // 256E (vm)
    float* attn  = ws + 1574 * E; // 8E
    // node-scale buffers
    float* nb    = ws + 1582 * E;
    float* x     = nb;                 // 256N
    float* vec   = nb + 256 * Nn;      // 768N
    float* xn    = nb + 1024 * Nn;     // 256N
    float* vecn  = nb + 1280 * Nn;     // 768N
    float* q     = nb + 2048 * Nn;     // 256N
    float* k     = nb + 2304 * Nn;     // 256N
    float* v     = nb + 2560 * Nn;     // 256N
    float* vecp  = nb + 2816 * Nn;     // 2304N
    float* vdot  = nb + 5120 * Nn;     // 256N
    float* xagg  = nb + 5376 * Nn;     // 256N
    float* vagg  = nb + 5632 * Nn;     // 768N
    float* o     = nb + 6400 * Nn;     // 768N
    float* wt    = nb + 7168 * Nn;     // 3072N
    float* xcat  = nb + 10240 * Nn;    // 512N

    auto gemm = [&](const float* A, const float* B, const float* bias, float* C,
                    int M, int K, int Nc, bool silu_act) {
        dim3 grid(M / 128, Nc / 64);
        if (silu_act)
            gemm_wmma_kernel<1><<<grid, 256, 0, stream>>>(A, B, bias, C, M, K, Nc);
        else
            gemm_wmma_kernel<0><<<grid, 256, 0, stream>>>(A, B, bias, C, M, K, Nc);
    };

    const int TB = 256;
    const int gEH  = (int)((E * HDIM) / TB);        // 65536
    const int gNH  = (int)((Nn * HDIM) / TB);       // 4096
    const int gN3H = (int)((Nn * 3 * HDIM) / TB);   // 12288
    const int gE   = (int)(E / TB);                 // 256
    const int gEH8 = (int)((E * NHEADS) / TB);      // 2048

    // ---- edge geometry + RBF ----
    edge_geom_kernel<<<gE, TB, 0, stream>>>(pos, ei, r, Cc, mask, dij, earbf);

    // ---- NeighborEmbedding ----
    gemm(earbf, dist_w, dist_b, be3, NEDGE, NRBF, HDIM, false);   // W
    hipMemsetAsync(xagg, 0, (size_t)256 * Nn * sizeof(float), stream);
    neighbor_agg_kernel<<<gEH, TB, 0, stream>>>(be3, Cc, mask, ei, z, nbr, xagg);
    xcat_kernel<<<gNH, TB, 0, stream>>>(z, emb, xagg, xcat);
    gemm(xcat, comb_w, comb_b, x, NNODE, 2 * HDIM, HDIM, false);  // x

    // ---- EdgeEmbedding ----
    gemm(earbf, edge_w, edge_b, be3, NEDGE, NRBF, HDIM, false);
    eatt_init_kernel<<<gEH, TB, 0, stream>>>(x, ei, be3, eatt);

    // ---- vec = 0 ----
    hipMemsetAsync(vec, 0, (size_t)768 * Nn * sizeof(float), stream);

    // ---- ViS-MP layers ----
    for (int l = 0; l < NLAYER; ++l) {
        const float* awl = aw + (long long)l * 5 * HDIM * HDIM;
        const float* abl = ab + (long long)l * 5 * HDIM;

        layernorm_kernel<<<(int)(Nn / 8), TB, 0, stream>>>(x, ln_g + l * HDIM, ln_b + l * HDIM, xn, NNODE);
        vecn_kernel<<<gN3H, TB, 0, stream>>>(vec, vln_w + l * HDIM, vecn);

        gemm(xn, awl + 0 * HDIM * HDIM, abl + 0 * HDIM, q, NNODE, HDIM, HDIM, false);
        gemm(xn, awl + 1 * HDIM * HDIM, abl + 1 * HDIM, k, NNODE, HDIM, HDIM, false);
        gemm(xn, awl + 2 * HDIM * HDIM, abl + 2 * HDIM, v, NNODE, HDIM, HDIM, false);
        gemm(eatt, awl + 3 * HDIM * HDIM, abl + 3 * HDIM, be1, NEDGE, HDIM, HDIM, true);  // dk
        gemm(eatt, awl + 4 * HDIM * HDIM, abl + 4 * HDIM, be2, NEDGE, HDIM, HDIM, true);  // dv
        gemm(vecn, vpw + (long long)l * HDIM * 3 * HDIM, nullptr, vecp, 3 * NNODE, HDIM, 3 * HDIM, false);

        vecdot_kernel<<<gNH, TB, 0, stream>>>(vecp, vdot);
        attn_kernel<<<gEH8, TB, 0, stream>>>(q, k, be1, Cc, ei, attn);
        vm_kernel<<<gEH, TB, 0, stream>>>(v, be2, attn, ei, be4);

        gemm(be4, sw + (long long)l * HDIM * 2 * HDIM, sb + (long long)l * 2 * HDIM,
             be3, NEDGE, HDIM, 2 * HDIM, true);                                           // s

        hipMemsetAsync(xagg, 0, (size_t)256 * Nn * sizeof(float), stream);
        hipMemsetAsync(vagg, 0, (size_t)768 * Nn * sizeof(float), stream);
        scatter_kernel<<<gEH, TB, 0, stream>>>(be4, be3, vecn, dij, ei, xagg, vagg);

        gemm(xagg, ow + (long long)l * HDIM * 3 * HDIM, ob + (long long)l * 3 * HDIM,
             o, NNODE, HDIM, 3 * HDIM, false);
        update_kernel<<<gNH, TB, 0, stream>>>(o, vdot, vecp, vagg, x, vec);

        if (l < NLAYER - 1) {
            const float* wtl = wtw + (long long)l * 4 * HDIM * HDIM;
            for (int p = 0; p < 4; ++p)
                gemm(vecn, wtl + (long long)p * HDIM * HDIM, nullptr,
                     wt + (long long)p * 3 * Nn * HDIM, 3 * NNODE, HDIM, HDIM, false);
            dots_kernel<<<gEH, TB, 0, stream>>>(wt, ei, be1, be2);
            gemm(eatt, fw + (long long)l * HDIM * 2 * HDIM, fb + (long long)l * 2 * HDIM,
                 be3, NEDGE, HDIM, 2 * HDIM, true);                                       // f
            eatt_update_kernel<<<gEH, TB, 0, stream>>>(be3, be1, be2, eatt);
        }
    }

    // ---- outputs ----
    layernorm_kernel<<<(int)(Nn / 8), TB, 0, stream>>>(x, outg, outb, out, NNODE);
    vout_kernel<<<gN3H, TB, 0, stream>>>(vec, voutw, out);
}
